// TemporalAttentionLayer_43946105373022
// MI455X (gfx1250) — matrix operand order repaired and checked by reading, not compiled, not run on hardware
//
#include <hip/hip_runtime.h>
#include <cstdint>

typedef uint16_t u16;
typedef __attribute__((ext_vector_type(16))) __bf16 v16bf;
typedef __attribute__((ext_vector_type(8)))  __bf16 v8bf;
typedef __attribute__((ext_vector_type(8)))  float  v8f;

#define D_MODEL 2048
#define NFRAMES 4
#define STOK    768
#define LOTHER  48
#define SEQLEN  (NFRAMES*STOK + LOTHER)   // 3120
#define BATCH   4
#define MTOK    (BATCH*STOK*NFRAMES)      // 12288
#define HDIM    128

#define BM 128
#define BN 128
#define BK 32
#define LDT 40   // padded LDS row stride in halves (80B -> conflict-free b128 gathers)

__device__ __forceinline__ u16 f2bf(float f) {
  uint32_t u = __float_as_uint(f);
  u += 0x7FFFu + ((u >> 16) & 1u);     // round-to-nearest-even
  return (u16)(u >> 16);
}
__device__ __forceinline__ float bf2f(u16 h) {
  return __uint_as_float(((uint32_t)h) << 16);
}

// 16B async copy global -> LDS (tracked by ASYNCcnt, no VGPR staging).
// SADDR form: uniform 64-bit base (SGPR pair) + per-lane 32-bit byte offset.
__device__ __forceinline__ void async_ld16(uint32_t lds_byte_off, uint32_t glb_byte_off,
                                           const void* sbase) {
  asm volatile("global_load_async_to_lds_b128 %0, %1, %2"
               :
               : "v"(lds_byte_off), "v"(glb_byte_off), "s"(sbase)
               : "memory");
}
__device__ __forceinline__ void wait_async0() {
#if __has_builtin(__builtin_amdgcn_s_wait_asynccnt)
  __builtin_amdgcn_s_wait_asynccnt(0);
#else
  asm volatile("s_wait_asynccnt 0x0" ::: "memory");
#endif
}
__device__ __forceinline__ uint32_t lds_off32(const void* p) {
  // flat address of a __shared__ object: high 32 = LDS aperture, low 32 = LDS byte offset
  return (uint32_t)(uintptr_t)p;
}

// ---------------------------------------------------------------- weights f32 -> bf16
__global__ void convert_bf16_kernel(const float* __restrict__ src, u16* __restrict__ dst, int n) {
  int i = blockIdx.x * blockDim.x + threadIdx.x;
  if (i < n) dst[i] = f2bf(src[i]);
}

// ---------------------------------------------------------------- LayerNorm + pos-enc -> Xb[(b*S+s)*N+n][d] bf16
__global__ __launch_bounds__(256) void ln_pe_kernel(const float* __restrict__ hs,
                                                    const float* __restrict__ gamma,
                                                    const float* __restrict__ beta,
                                                    const float* __restrict__ fintv,
                                                    u16* __restrict__ Xb) {
  __shared__ float red1[256];
  __shared__ float red2[256];
  int t = blockIdx.x;                       // (b, n, s) order over image tokens
  int b = t / (NFRAMES * STOK);
  int r = t % (NFRAMES * STOK);
  int n = r / STOK;
  int s = r % STOK;
  const float* row = hs + ((size_t)b * SEQLEN + (size_t)n * STOK + s) * D_MODEL;

  float x[8];
  float sum = 0.f, sq = 0.f;
#pragma unroll
  for (int i = 0; i < 8; ++i) {
    x[i] = row[threadIdx.x + 256 * i];
    sum += x[i];
    sq  += x[i] * x[i];
  }
  red1[threadIdx.x] = sum;
  red2[threadIdx.x] = sq;
  __syncthreads();
  for (int o = 128; o > 0; o >>= 1) {
    if ((int)threadIdx.x < o) {
      red1[threadIdx.x] += red1[threadIdx.x + o];
      red2[threadIdx.x] += red2[threadIdx.x + o];
    }
    __syncthreads();
  }
  float mean = red1[0] * (1.f / D_MODEL);
  float var  = red2[0] * (1.f / D_MODEL) - mean * mean;
  float rstd = rsqrtf(var + 1e-5f);
  float time = (float)n * fintv[0];

  u16* orow = Xb + (((size_t)(b * STOK + s)) * NFRAMES + n) * D_MODEL;
#pragma unroll
  for (int i = 0; i < 8; ++i) {
    int d = threadIdx.x + 256 * i;
    int j = (d < 1024) ? d : (d - 1024);
    // scaling = 2*pi / (0.01 * 1000^(j/1023))
    float sc  = 628.3185307f * __expf(-(float)j * (6.90775527898f / 1023.0f));
    float ang = time * sc;
    float pe  = (d < 1024) ? __sinf(ang) : __cosf(ang);
    float val = (x[i] - mean) * rstd * gamma[d] + beta[d] + pe;
    orow[d] = f2bf(val);
  }
}

// ---------------------------------------------------------------- bf16 WMMA GEMM: C[M,2048] = A[M,2048] * W[2048,2048]^T
// Async-staged (GLOBAL_LOAD_ASYNC_TO_LDS_B128), LDS double-buffered via XOR byte toggle,
// K loop kept rolled so accumulator PHIs coalesce with in-place WMMA D=C tuples.
// MODE 0: store bf16 row-major (Q/K/V, attention-out projection input)
// MODE 1: residual add + scatter-store f32 into (B, N*S, D) layout of d_out
template <int MODE>
__global__ __launch_bounds__(256) void gemm_kernel(const u16* __restrict__ A,
                                                   const u16* __restrict__ W,
                                                   u16* __restrict__ Cb,
                                                   float* __restrict__ Cf,
                                                   const float* __restrict__ resid) {
  __shared__ alignas(16) u16 As[2 * BM * LDT];
  __shared__ alignas(16) u16 Bs[2 * BN * LDT];
  const uint32_t TOGB = (uint32_t)(BM * LDT * 2);   // bytes between the two buffers

  const int tid   = threadIdx.x;
  const int lane  = tid & 31;
  const int wave  = tid >> 5;
  const int waveM = (wave >> 2) * 64;   // 2x4 wave grid over the 128x128 block
  const int waveN = (wave & 3) * 32;
  const int blockN = blockIdx.x * BN;
  const int blockM = blockIdx.y * BM;

  // staging map: thread t owns 32B of one row (row = t/2, 16-half chunk = t&1)
  const int lrow = tid >> 1;
  const int lcol = (tid & 1) * 16;
  const uint32_t aGlb = (uint32_t)(((blockM + lrow) * D_MODEL + lcol) * 2);  // bytes
  const uint32_t bGlb = (uint32_t)(((blockN + lrow) * D_MODEL + lcol) * 2);  // bytes
  const uint32_t ldsA0 = lds_off32(&As[lrow * LDT + lcol]);
  const uint32_t ldsB0 = lds_off32(&Bs[lrow * LDT + lcol]);

  auto issueTile = [&](uint32_t bufB, uint32_t kb) {
    async_ld16(ldsA0 + bufB,      aGlb + kb,      A);
    async_ld16(ldsA0 + bufB + 16, aGlb + kb + 16, A);
    async_ld16(ldsB0 + bufB,      bGlb + kb,      W);
    async_ld16(ldsB0 + bufB + 16, bGlb + kb + 16, W);
  };

  v8f acc[4][2];
#pragma unroll
  for (int i = 0; i < 4; ++i)
#pragma unroll
    for (int j = 0; j < 2; ++j)
      acc[i][j] = (v8f){0.f, 0.f, 0.f, 0.f, 0.f, 0.f, 0.f, 0.f};

  // per-lane fragment gather bases (ISA 16-bit A 16x32 / B 32x16 layouts)
  const int hsel = lane >> 4;
  const int aIdx = (waveM + (lane & 15)) * LDT + hsel * 8;
  const int bIdx = (waveN + (lane & 15)) * LDT + hsel * 8;

  union Frag { v16bf v; v8bf h[2]; };

  issueTile(0, 0);
  wait_async0();
  __syncthreads();

  const int KSTEPS = D_MODEL / BK;   // 64
  uint32_t off = 0;                  // byte toggle selecting the compute buffer
#pragma unroll 1
  for (int kt = 0; kt < KSTEPS; ++kt) {
    if (kt + 1 < KSTEPS) issueTile(off ^ TOGB, (uint32_t)(kt + 1) << 6);

    const u16* aB = As + (off >> 1) + aIdx;
    const u16* bB = Bs + (off >> 1) + bIdx;

    Frag bfr[2];
#pragma unroll
    for (int tj = 0; tj < 2; ++tj) {
      const u16* p = bB + tj * 16 * LDT;
      bfr[tj].h[0] = *reinterpret_cast<const v8bf*>(p);
      bfr[tj].h[1] = *reinterpret_cast<const v8bf*>(p + 16);
    }
#pragma unroll
    for (int ti = 0; ti < 4; ++ti) {
      const u16* p = aB + ti * 16 * LDT;
      Frag afr;
      afr.h[0] = *reinterpret_cast<const v8bf*>(p);
      afr.h[1] = *reinterpret_cast<const v8bf*>(p + 16);
#pragma unroll
      for (int tj = 0; tj < 2; ++tj)
        acc[ti][tj] = __builtin_amdgcn_wmma_f32_16x16x32_bf16(
            false, afr.v, false, bfr[tj].v, (short)0, acc[ti][tj], false, false);
    }

    wait_async0();       // my async stores into the other buffer are in LDS
    __syncthreads();     // everyone finished reads of this buffer + their loads
    off ^= TOGB;
  }

  // epilogue (C/D layout: VGPR r -> M = r + 8*(lane/16), N = lane%16)
#pragma unroll
  for (int ti = 0; ti < 4; ++ti)
#pragma unroll
    for (int tj = 0; tj < 2; ++tj)
#pragma unroll
      for (int r = 0; r < 8; ++r) {
        int grow = blockM + waveM + ti * 16 + r + (hsel << 3);
        int gcol = blockN + waveN + tj * 16 + (lane & 15);
        float v = acc[ti][tj][r];
        if (MODE == 0) {
          Cb[(size_t)grow * D_MODEL + gcol] = f2bf(v);
        } else {
          int b  = grow / (STOK * NFRAMES);
          int rm = grow % (STOK * NFRAMES);
          int s  = rm >> 2;            // N = 4 frames
          int n  = rm & 3;
          size_t dst = ((size_t)b * SEQLEN + (size_t)n * STOK + s) * D_MODEL + gcol;
          Cf[dst] = resid[dst] + v;
        }
      }
}

// ---------------------------------------------------------------- temporal attention (4x4 per head) over frames
__global__ __launch_bounds__(256) void attn_kernel(const u16* __restrict__ Q,
                                                   const u16* __restrict__ K,
                                                   const u16* __restrict__ V,
                                                   u16* __restrict__ AO) {
  const int bs   = blockIdx.x;        // b*S + s
  const int lane = threadIdx.x & 31;
  const int wave = threadIdx.x >> 5;
  const float scaling = 0.08838834764831845f;   // 1/sqrt(128)

  for (int hh = 0; hh < 2; ++hh) {
    int h = wave * 2 + hh;
    size_t base = (size_t)bs * NFRAMES * D_MODEL + (size_t)h * HDIM + (size_t)lane * 4;

    float q[4][4], k[4][4], v[4][4];
#pragma unroll
    for (int n = 0; n < 4; ++n) {
      const u16* qp = Q + base + (size_t)n * D_MODEL;
      const u16* kp = K + base + (size_t)n * D_MODEL;
      const u16* vp = V + base + (size_t)n * D_MODEL;
#pragma unroll
      for (int j = 0; j < 4; ++j) {
        q[n][j] = bf2f(qp[j]);
        k[n][j] = bf2f(kp[j]);
        v[n][j] = bf2f(vp[j]);
      }
    }

    float p[4][4];
#pragma unroll
    for (int n = 0; n < 4; ++n)
#pragma unroll
      for (int m = 0; m < 4; ++m) {
        float d = q[n][0] * k[m][0] + q[n][1] * k[m][1] + q[n][2] * k[m][2] + q[n][3] * k[m][3];
        for (int o = 16; o > 0; o >>= 1) d += __shfl_xor(d, o, 32);
        p[n][m] = d * scaling;
      }

#pragma unroll
    for (int n = 0; n < 4; ++n) {
      float mx = fmaxf(fmaxf(p[n][0], p[n][1]), fmaxf(p[n][2], p[n][3]));
      float e0 = __expf(p[n][0] - mx), e1 = __expf(p[n][1] - mx);
      float e2 = __expf(p[n][2] - mx), e3 = __expf(p[n][3] - mx);
      float inv = 1.f / (e0 + e1 + e2 + e3);
      p[n][0] = e0 * inv; p[n][1] = e1 * inv; p[n][2] = e2 * inv; p[n][3] = e3 * inv;
    }

#pragma unroll
    for (int n = 0; n < 4; ++n) {
      u16* op = AO + base + (size_t)n * D_MODEL;
#pragma unroll
      for (int j = 0; j < 4; ++j) {
        float o = p[n][0] * v[0][j] + p[n][1] * v[1][j] + p[n][2] * v[2][j] + p[n][3] * v[3][j];
        op[j] = f2bf(o);
      }
    }
  }
}

// ---------------------------------------------------------------- passthrough of the "other" tokens
__global__ void copy_other_kernel(const float* __restrict__ hs, float* __restrict__ out) {
  int i = blockIdx.x * blockDim.x + threadIdx.x;
  const int total = BATCH * LOTHER * D_MODEL;
  if (i < total) {
    int b   = i / (LOTHER * D_MODEL);
    int rem = i % (LOTHER * D_MODEL);
    size_t idx = ((size_t)b * SEQLEN + (size_t)(NFRAMES * STOK)) * D_MODEL + rem;
    out[idx] = hs[idx];
  }
}

extern "C" void kernel_launch(void* const* d_in, const int* in_sizes, int n_in,
                              void* d_out, int out_size, void* d_ws, size_t ws_size,
                              hipStream_t stream) {
  (void)in_sizes; (void)n_in; (void)out_size; (void)ws_size;
  const float* hs    = (const float*)d_in[0];
  const float* Wq    = (const float*)d_in[1];
  const float* Wk    = (const float*)d_in[2];
  const float* Wv    = (const float*)d_in[3];
  const float* Wo    = (const float*)d_in[4];
  const float* gamma = (const float*)d_in[5];
  const float* beta  = (const float*)d_in[6];
  const float* fintv = (const float*)d_in[9];
  float* out = (float*)d_out;

  u16* ws = (u16*)d_ws;
  const size_t tokElems = (size_t)MTOK * D_MODEL;        // 25.2M
  const size_t wElems   = (size_t)D_MODEL * D_MODEL;     // 4.2M
  u16* Xb  = ws;
  u16* Qb  = Xb + tokElems;
  u16* Kb  = Qb + tokElems;
  u16* Vb  = Kb + tokElems;
  u16* Ab  = Vb + tokElems;
  u16* Wqb = Ab + tokElems;
  u16* Wkb = Wqb + wElems;
  u16* Wvb = Wkb + wElems;
  u16* Wob = Wvb + wElems;

  const int wN = (int)wElems;
  const int cblk = (wN + 255) / 256;
  convert_bf16_kernel<<<cblk, 256, 0, stream>>>(Wq, Wqb, wN);
  convert_bf16_kernel<<<cblk, 256, 0, stream>>>(Wk, Wkb, wN);
  convert_bf16_kernel<<<cblk, 256, 0, stream>>>(Wv, Wvb, wN);
  convert_bf16_kernel<<<cblk, 256, 0, stream>>>(Wo, Wob, wN);

  ln_pe_kernel<<<MTOK, 256, 0, stream>>>(hs, gamma, beta, fintv, Xb);

  dim3 gg(D_MODEL / BN, MTOK / BM);   // 16 x 96
  gemm_kernel<0><<<gg, 256, 0, stream>>>(Xb, Wqb, Qb, nullptr, nullptr);
  gemm_kernel<0><<<gg, 256, 0, stream>>>(Xb, Wkb, Kb, nullptr, nullptr);
  gemm_kernel<0><<<gg, 256, 0, stream>>>(Xb, Wvb, Vb, nullptr, nullptr);

  attn_kernel<<<BATCH * STOK, 256, 0, stream>>>(Qb, Kb, Vb, Ab);

  gemm_kernel<1><<<gg, 256, 0, stream>>>(Ab, Wob, nullptr, out, hs);

  const int otherN = BATCH * LOTHER * D_MODEL;
  copy_other_kernel<<<(otherN + 255) / 256, 256, 0, stream>>>(hs, out);
}
